// Mamba2HSIClassifier_79585743995308
// MI455X (gfx1250) — compile-verified, matrix operand order checked by reading, not run on hardware
//
#include <hip/hip_runtime.h>
#include <hip/hip_bf16.h>
#include <math.h>

// ---------------------------------------------------------------------------
// Model constants
// ---------------------------------------------------------------------------
#define IMGX      145
#define LSEQ      (IMGX * IMGX)      // 21025
#define BANDS     200
#define KB_PAD    224                // 200 padded to mult of 32
#define DMODEL    256
#define DINNER    512
#define DSTATE    128
#define NHEADS    8
#define HEADDIM   64
#define DCONV     4
#define CHUNK     256
#define DXBC      (DINNER + 2 * DSTATE)              // 768
#define DINPROJ   (2 * DINNER + 2 * DSTATE + NHEADS) // 1288
#define ZLD       1312               // 1288 padded to mult of 32
#define NCLS      17
#define GCHUNKS   83
#define LPAD      (GCHUNKS * CHUNK)  // 21248 = 128*166 (all tiles full)

typedef __attribute__((ext_vector_type(16))) __bf16        v16bf;
typedef __attribute__((ext_vector_type(8)))  float         v8f;
typedef __attribute__((ext_vector_type(8)))  unsigned int  v8u;

// ---------------------------------------------------------------------------
// Device helpers
// ---------------------------------------------------------------------------
__device__ __forceinline__ __bf16 f2bf(float f) {
  unsigned u = __builtin_bit_cast(unsigned, f);
  u += 0x7FFFu + ((u >> 16) & 1u);                 // round-to-nearest-even
  unsigned short s = (unsigned short)(u >> 16);
  return __builtin_bit_cast(__bf16, s);
}
__device__ __forceinline__ float geluf(float x) {
  return 0.5f * x * (1.0f + erff(x * 0.70710678118654752440f));
}
__device__ __forceinline__ float siluf(float x) {
  return x / (1.0f + __expf(-x));
}
__device__ __forceinline__ float wave_sum(float v) {
#pragma unroll
  for (int o = 16; o > 0; o >>= 1) v += __shfl_xor(v, o, 32);
  return v;
}
// bf16 fragment load, CDNA5 16-bit WMMA A layout. p = row base + k0 + laneHalf.
// Per lane: elements k+{0..7} and k+{16..23} -> two b128 loads, zero ALU.
__device__ __forceinline__ v16bf ld_fragbf(const __bf16* __restrict__ p) {
  uint4 a = *(const uint4*)(p);
  uint4 b = *(const uint4*)(p + 16);
  v8u u;
  u[0] = a.x; u[1] = a.y; u[2] = a.z; u[3] = a.w;
  u[4] = b.x; u[5] = b.y; u[6] = b.z; u[7] = b.w;
  return __builtin_bit_cast(v16bf, u);
}
__device__ __forceinline__ v16bf ld_fragbf_mask(const __bf16* __restrict__ p,
                                                unsigned msk) {
  uint4 a = *(const uint4*)(p);
  uint4 b = *(const uint4*)(p + 16);
  v8u u;
  u[0] = a.x & msk; u[1] = a.y & msk; u[2] = a.z & msk; u[3] = a.w & msk;
  u[4] = b.x & msk; u[5] = b.y & msk; u[6] = b.z & msk; u[7] = b.w & msk;
  return __builtin_bit_cast(v16bf, u);
}
__device__ __forceinline__ v8f wmma_bf16(v16bf a, v16bf b, v8f c) {
  return __builtin_amdgcn_wmma_f32_16x16x32_bf16(false, a, false, b,
                                                 (short)0, c, false, false);
}

// ---------------------------------------------------------------------------
// Generic WMMA GEMM, 2x2 tiles (32x32) per wave, 4 waves/block along M.
// All operands bf16, K % 32 == 0, all tile rows addressable (padded buffers).
// Y[m,n] = act(sum_k X[m,k]*W[n,k] + bias[n]); fp32 or bf16 output.
// ---------------------------------------------------------------------------
__global__ void k_gemm(const __bf16* __restrict__ X, long xRs, long xBs,
                       const __bf16* __restrict__ W, long wRs, long wBs,
                       const float* __restrict__ bias,
                       void* __restrict__ Yv, long yLd, long yBs,
                       int Mst, int Nst, int K, int act, int bfout) {
  int lane = threadIdx.x & 31;
  long n0 = (long)blockIdx.x * 32;
  long m0 = ((long)blockIdx.y * 4 + (threadIdx.x >> 5)) * 32;
  int b = blockIdx.z;
  int r = lane & 15, kofs = (lane & 16) ? 8 : 0;
  const __bf16* xr0 = X + (long)b * xBs + (m0 + r) * xRs + kofs;
  const __bf16* xr1 = xr0 + 16 * xRs;
  const __bf16* wr0 = W + (long)b * wBs + (n0 + r) * wRs + kofs;
  const __bf16* wr1 = wr0 + 16 * wRs;
  v8f a00 = {}, a01 = {}, a10 = {}, a11 = {};
  for (int k0 = 0; k0 < K; k0 += 32) {
    v16bf xa = ld_fragbf(xr0 + k0);
    v16bf xb = ld_fragbf(xr1 + k0);
    v16bf wa = ld_fragbf(wr0 + k0);
    v16bf wb = ld_fragbf(wr1 + k0);
    a00 = wmma_bf16(xa, wa, a00);
    a01 = wmma_bf16(xa, wb, a01);
    a10 = wmma_bf16(xb, wa, a10);
    a11 = wmma_bf16(xb, wb, a11);
  }
  int mhalf = (lane & 16) ? 8 : 0;
#pragma unroll
  for (int nt = 0; nt < 2; ++nt) {
    long n = n0 + nt * 16 + r;
    if (n >= Nst) continue;
    float bv = bias ? bias[n] : 0.0f;
#pragma unroll
    for (int mt = 0; mt < 2; ++mt) {
      v8f A = nt ? (mt ? a11 : a01) : (mt ? a10 : a00);
      long mb = m0 + mt * 16 + mhalf;
#pragma unroll
      for (int j = 0; j < 8; ++j) {
        long m = mb + j;
        if (m < Mst) {
          float v = A[j] + bv;
          if (act) v = geluf(v);
          if (bfout) ((__bf16*)Yv)[(long)b * yBs + m * yLd + n] = f2bf(v);
          else       ((float*)Yv)[(long)b * yBs + m * yLd + n] = v;
        }
      }
    }
  }
}

// ---------------------------------------------------------------------------
// 3x3 SAME conv: all 9 taps accumulated in-register, bias+GELU fused, bf16 out.
// xn: bf16 (LPAD,224); WT: bf16 (9,256,224); Y: bf16 (LPAD,256).
// Invalid shifted rows: address clamped to own row, value masked via AND.
// ---------------------------------------------------------------------------
__global__ void k_conv(const __bf16* __restrict__ xn,
                       const __bf16* __restrict__ WT,
                       const float* __restrict__ bias,
                       __bf16* __restrict__ Y) {
  int lane = threadIdx.x & 31;
  long n0 = (long)blockIdx.x * 32;
  long m0 = ((long)blockIdx.y * 4 + (threadIdx.x >> 5)) * 32;
  int r = lane & 15, kofs = (lane & 16) ? 8 : 0;
  long mA = m0 + r, mB = mA + 16;
  int rA = (int)(mA / IMGX), cA = (int)(mA % IMGX);
  int rB = (int)(mB / IMGX), cB = (int)(mB % IMGX);
  bool inA = mA < LSEQ, inB = mB < LSEQ;
  v8f a00 = {}, a01 = {}, a10 = {}, a11 = {};
  for (int tap = 0; tap < 9; ++tap) {
    int dy = tap / 3 - 1, dx = tap % 3 - 1;
    int rrA = rA + dy, ccA = cA + dx;
    int rrB = rB + dy, ccB = cB + dx;
    bool vA = inA && rrA >= 0 && rrA < IMGX && ccA >= 0 && ccA < IMGX;
    bool vB = inB && rrB >= 0 && rrB < IMGX && ccB >= 0 && ccB < IMGX;
    long rowA = vA ? (long)(rrA * IMGX + ccA) : mA;
    long rowB = vB ? (long)(rrB * IMGX + ccB) : mB;
    unsigned mkA = vA ? 0xFFFFFFFFu : 0u, mkB = vB ? 0xFFFFFFFFu : 0u;
    const __bf16* pa = xn + rowA * KB_PAD + kofs;
    const __bf16* pb = xn + rowB * KB_PAD + kofs;
    const __bf16* w0 = WT + (long)tap * DMODEL * KB_PAD + (n0 + r) * KB_PAD + kofs;
    const __bf16* w1 = w0 + 16 * KB_PAD;
    for (int k0 = 0; k0 < KB_PAD; k0 += 32) {
      v16bf xa = ld_fragbf_mask(pa + k0, mkA);
      v16bf xb = ld_fragbf_mask(pb + k0, mkB);
      v16bf wa = ld_fragbf(w0 + k0);
      v16bf wb = ld_fragbf(w1 + k0);
      a00 = wmma_bf16(xa, wa, a00);
      a01 = wmma_bf16(xa, wb, a01);
      a10 = wmma_bf16(xb, wa, a10);
      a11 = wmma_bf16(xb, wb, a11);
    }
  }
  int mhalf = (lane & 16) ? 8 : 0;
#pragma unroll
  for (int nt = 0; nt < 2; ++nt) {
    long n = n0 + nt * 16 + r;
    float bv = bias[n];
#pragma unroll
    for (int mt = 0; mt < 2; ++mt) {
      v8f A = nt ? (mt ? a11 : a01) : (mt ? a10 : a00);
      long mb = m0 + mt * 16 + mhalf;
#pragma unroll
      for (int j = 0; j < 8; ++j)
        Y[(mb + j) * DMODEL + n] = f2bf(geluf(A[j] + bv));
    }
  }
}

// ---------------------------------------------------------------------------
// Weight pad/transpose copy to bf16: dst(Npad,Kpad) <- src(n,k), 0 outside.
// ---------------------------------------------------------------------------
__global__ void k_padw(const float* __restrict__ src, long sRs, long sKs, long sBs,
                       __bf16* __restrict__ dst, int Npad, int Kpad, long dBs,
                       int N, int K) {
  long i = blockIdx.x * (long)blockDim.x + threadIdx.x;
  if (i >= (long)Npad * Kpad) return;
  int b = blockIdx.y;
  int n = (int)(i / Kpad), k = (int)(i % Kpad);
  float v = (n < N && k < K) ? src[(long)n * sRs + (long)k * sKs + (long)b * sBs] : 0.f;
  dst[(long)b * dBs + i] = f2bf(v);
}
__global__ void k_padb(const float* __restrict__ src, float* __restrict__ dst,
                       int N, int Npad) {
  int i = blockIdx.x * blockDim.x + threadIdx.x;
  if (i < Npad) dst[i] = (i < N) ? src[i] : 0.f;
}

// ---------------------------------------------------------------------------
// LayerNorm (+ optional GELU), one wave per row, bf16 out at column offset.
// ---------------------------------------------------------------------------
__global__ void k_ln(const float* __restrict__ src, int lds,
                     const float* __restrict__ g, const float* __restrict__ b,
                     __bf16* __restrict__ dst, int ldd, int dofs,
                     int rows, int C, int do_gelu) {
  int row = blockIdx.x;
  if (row >= rows) return;
  int lane = threadIdx.x & 31;
  const float* p = src + (long)row * lds;
  float s = 0.f;
  for (int c = lane; c < C; c += 32) s += p[c];
  float mean = wave_sum(s) / (float)C;
  float v = 0.f;
  for (int c = lane; c < C; c += 32) { float d = p[c] - mean; v += d * d; }
  v = wave_sum(v) / (float)C;
  float rs = rsqrtf(v + 1e-5f);
  __bf16* q = dst + (long)row * ldd + dofs;
  for (int c = lane; c < C; c += 32) {
    float y = (p[c] - mean) * rs * g[c] + b[c];
    if (do_gelu) y = geluf(y);
    q[c] = f2bf(y);
  }
}

// dt = softplus(zx[:,1280+h] + dt_bias[h]); zero for padded rows.
__global__ void k_dt(const float* __restrict__ zx,
                     const float* __restrict__ dt_bias,
                     float* __restrict__ dt) {
  int i = blockIdx.x * blockDim.x + threadIdx.x;
  if (i >= LPAD * NHEADS) return;
  int l = i >> 3, h = i & 7;
  float v = 0.f;
  if (l < LSEQ) {
    float x = zx[(long)l * ZLD + 2 * DINNER + 2 * DSTATE + h] + dt_bias[h];
    v = (x > 20.f) ? x : log1pf(__expf(x));
  }
  dt[i] = v;
}

// Depthwise causal conv over sequence + SiLU; writes fp32 + bf16 copies.
__global__ void k_convseq(const float* __restrict__ zx,
                          const float* __restrict__ cw,
                          const float* __restrict__ cb,
                          float* __restrict__ xbc,
                          __bf16* __restrict__ xbcb) {
  long i = blockIdx.x * (long)blockDim.x + threadIdx.x;
  if (i >= (long)LPAD * DXBC) return;
  int l = (int)(i / DXBC), c = (int)(i % DXBC);
  float v = 0.f;
  if (l < LSEQ) {
    float s = cb[c];
#pragma unroll
    for (int k = 0; k < DCONV; ++k) {
      int ls = l - (DCONV - 1) + k;
      if (ls >= 0) s += zx[(long)ls * ZLD + DINNER + c] * cw[c * DCONV + k];
    }
    v = siluf(s);
  }
  xbc[i] = v;
  xbcb[i] = f2bf(v);
}

// Per-(chunk,head) inclusive cumsum of a = dt * (-exp(A_log[h])):
// acsT[(g*8+h)*256 + i].
__global__ void k_cumsum(const float* __restrict__ dt,
                         const float* __restrict__ A_log,
                         float* __restrict__ acsT) {
  __shared__ float sh[CHUNK];
  int g = blockIdx.x >> 3, h = blockIdx.x & 7;
  int i = threadIdx.x;
  float A = -__expf(A_log[h]);
  sh[i] = dt[(g * CHUNK + i) * NHEADS + h] * A;
  __syncthreads();
  for (int o = 1; o < CHUNK; o <<= 1) {
    float t = (i >= o) ? sh[i - o] : 0.f;
    __syncthreads();
    sh[i] += t;
    __syncthreads();
  }
  acsT[(long)blockIdx.x * CHUNK + i] = sh[i];
}

// dstT[gh,j] = exp(acsT[gh,255] - acsT[gh,j])
__global__ void k_dstate(const float* __restrict__ acsT, float* __restrict__ dstT) {
  int i = blockIdx.x * blockDim.x + threadIdx.x;
  if (i >= LPAD * NHEADS) return;
  int gh = i >> 8;
  dstT[i] = __expf(acsT[(long)gh * CHUNK + CHUNK - 1] - acsT[i]);
}

// xdtT[(gh*64+p)*256+j] = x[g*256+j,h,p]*dt[..]  (bf16),
// xds = xdtT * dstT[gh,j]                        (bf16, states A operand)
__global__ void k_xprep(const float* __restrict__ xbc, const float* __restrict__ dt,
                        const float* __restrict__ dstT,
                        __bf16* __restrict__ xdtT, __bf16* __restrict__ xds) {
  long i = blockIdx.x * (long)blockDim.x + threadIdx.x;
  if (i >= (long)LPAD * DINNER) return;
  int j = (int)(i & 255);
  long rw = i >> 8;
  int p = (int)(rw & 63);
  int gh = (int)(rw >> 6);
  long l = (long)(gh >> 3) * CHUNK + j;
  float v = xbc[l * DXBC + (gh & 7) * HEADDIM + p] * dt[l * NHEADS + (gh & 7)];
  xdtT[i] = f2bf(v);
  xds[i] = f2bf(v * dstT[(long)gh * CHUNK + j]);
}

// BT[(g*128+n)*256 + j] = B[g*256+j, n]   (bf16)
__global__ void k_BT(const float* __restrict__ xbc, __bf16* __restrict__ BT) {
  long i = blockIdx.x * (long)blockDim.x + threadIdx.x;
  if (i >= (long)LPAD * DSTATE) return;
  int j = (int)(i & 255);
  long rw = i >> 8;
  int n = (int)(rw & 127);
  int g = (int)(rw >> 7);
  BT[i] = f2bf(xbc[((long)g * CHUNK + j) * DXBC + DINNER + n]);
}

// y_diag: per (g,h): y[i,p] = sum_j (CB[i,j]*exp(acs_i-acs_j)*[j<=i]) * xdtT[p,j]
__global__ void k_ydiag(const float* __restrict__ CB,
                        const float* __restrict__ acsT,
                        const __bf16* __restrict__ xdtT,
                        float* __restrict__ y) {
  int gh = blockIdx.z, g = gh >> 3, h = gh & 7;
  int it = blockIdx.y * 4 + (threadIdx.x >> 5);  // 0..15
  int pt = blockIdx.x;                           // 0..3
  int lane = threadIdx.x & 31;
  int i = it * 16 + (lane & 15);
  int kofs = (lane & 16) ? 8 : 0;
  const float* acsr = acsT + (long)gh * CHUNK;
  float acs_i = acsr[i];
  const float* cbr = CB + (long)g * CHUNK * CHUNK + (long)i * CHUNK + kofs;
  const __bf16* xr = xdtT + ((long)gh * HEADDIM + pt * 16 + (lane & 15)) * CHUNK + kofs;
  v8f acc = {};
  for (int k0 = 0; k0 < CHUNK; k0 += 32) {
    const float4* cq = (const float4*)(cbr + k0);
    const float4* aq = (const float4*)(acsr + kofs + k0);
    float4 c0 = cq[0], c1 = cq[1], c2 = cq[4], c3 = cq[5];
    float4 d0 = aq[0], d1 = aq[1], d2 = aq[4], d3 = aq[5];
    float cbv[16] = {c0.x, c0.y, c0.z, c0.w, c1.x, c1.y, c1.z, c1.w,
                     c2.x, c2.y, c2.z, c2.w, c3.x, c3.y, c3.z, c3.w};
    float ajv[16] = {d0.x, d0.y, d0.z, d0.w, d1.x, d1.y, d1.z, d1.w,
                     d2.x, d2.y, d2.z, d2.w, d3.x, d3.y, d3.z, d3.w};
    int jb = k0 + kofs;
    v16bf a;
#pragma unroll
    for (int e = 0; e < 16; ++e) {
      int j = jb + (e < 8 ? e : 8 + e);
      float v = (j <= i) ? cbv[e] * __expf(acs_i - ajv[e]) : 0.f;  // exp<=1
      a[e] = f2bf(v);
    }
    v16bf bfr = ld_fragbf(xr + k0);
    acc = wmma_bf16(a, bfr, acc);
  }
  int p = pt * 16 + (lane & 15);
  long lb = (long)g * CHUNK + it * 16 + ((lane & 16) ? 8 : 0);
#pragma unroll
  for (int j = 0; j < 8; ++j)
    y[(lb + j) * DINNER + h * HEADDIM + p] = acc[j];
}

// states: per (g,h): st[p,n] = sum_j xds[p,j] * BT[n,j]
__global__ void k_states(const __bf16* __restrict__ xds,
                         const __bf16* __restrict__ BT,
                         float* __restrict__ st) {
  int gh = blockIdx.z, g = gh >> 3;
  int nt = blockIdx.x * 4 + (threadIdx.x >> 5);   // 0..7
  int pt = blockIdx.y;                            // 0..3
  int lane = threadIdx.x & 31;
  int kofs = (lane & 16) ? 8 : 0;
  const __bf16* xr = xds + ((long)gh * HEADDIM + pt * 16 + (lane & 15)) * CHUNK + kofs;
  const __bf16* br = BT + ((long)g * DSTATE + nt * 16 + (lane & 15)) * CHUNK + kofs;
  v8f acc = {};
  for (int k0 = 0; k0 < CHUNK; k0 += 32) {
    v16bf a = ld_fragbf(xr + k0);
    v16bf b = ld_fragbf(br + k0);
    acc = wmma_bf16(a, b, acc);
  }
  int n = nt * 16 + (lane & 15);
  long pb = pt * 16 + ((lane & 16) ? 8 : 0);
  float* S = st + (long)gh * HEADDIM * DSTATE;
#pragma unroll
  for (int j = 0; j < 8; ++j) S[(pb + j) * DSTATE + n] = acc[j];
}

// Sequential inter-chunk scan; sprev stored bf16 (yoff B operand).
__global__ void k_scan(const float* __restrict__ st,
                       const float* __restrict__ acsT,
                       __bf16* __restrict__ sprev) {
  int idx = blockIdx.x * blockDim.x + threadIdx.x;  // H*P*N = 65536
  int h = idx >> 13;
  float S = 0.f;
  for (int g = 0; g < GCHUNKS; ++g) {
    sprev[(long)g * 65536 + idx] = f2bf(S);
    float cd = __expf(acsT[((long)g * NHEADS + h) * CHUNK + CHUNK - 1]);
    S = cd * S + st[(long)g * 65536 + idx];
  }
}

// y += exp(acs_i) * (C @ Sprev^T) + D*x
__global__ void k_yoff(const __bf16* __restrict__ xbcb,
                       const float* __restrict__ xbc,
                       const __bf16* __restrict__ sprev,
                       const float* __restrict__ acsT,
                       const float* __restrict__ Dp,
                       float* __restrict__ y) {
  int gh = blockIdx.z, g = gh >> 3, h = gh & 7;
  int it = blockIdx.y * 4 + (threadIdx.x >> 5);   // 0..15
  int pt = blockIdx.x;                            // 0..3
  int lane = threadIdx.x & 31;
  int kofs = (lane & 16) ? 8 : 0;
  const __bf16* cr = xbcb + ((long)g * CHUNK + it * 16 + (lane & 15)) * DXBC
                     + DINNER + DSTATE + kofs;
  const __bf16* sr = sprev + (long)gh * HEADDIM * DSTATE
                     + (pt * 16 + (lane & 15)) * DSTATE + kofs;
  v8f acc = {};
  for (int k0 = 0; k0 < DSTATE; k0 += 32) {
    v16bf a = ld_fragbf(cr + k0);
    v16bf b = ld_fragbf(sr + k0);
    acc = wmma_bf16(a, b, acc);
  }
  int p = pt * 16 + (lane & 15);
  float D = Dp[h];
  const float* acsr = acsT + (long)gh * CHUNK;
  long ib = it * 16 + ((lane & 16) ? 8 : 0);
#pragma unroll
  for (int j = 0; j < 8; ++j) {
    long l = (long)g * CHUNK + ib + j;
    float e = __expf(acsr[ib + j]);
    long o = l * DINNER + h * HEADDIM + p;
    y[o] += e * acc[j] + D * xbc[l * DXBC + h * HEADDIM + p];
  }
}

// gated RMSNorm; bf16 out (out_proj A operand). One wave per row.
__global__ void k_gatenorm(const float* __restrict__ y,
                           const float* __restrict__ zx,
                           const float* __restrict__ nw,
                           __bf16* __restrict__ ybf) {
  int row = blockIdx.x;
  int lane = threadIdx.x & 31;
  const float* yr = y + (long)row * DINNER;
  const float* zr = zx + (long)row * ZLD;
  float vals[16];
  float ss = 0.f;
#pragma unroll
  for (int t = 0; t < 16; ++t) {
    int c = lane + t * 32;
    float v = yr[c] * siluf(zr[c]);
    vals[t] = v;
    ss += v * v;
  }
  ss = wave_sum(ss);
  float rs = rsqrtf(ss / (float)DINNER + 1e-5f);
  __bf16* qb = ybf + (long)row * DINNER;
#pragma unroll
  for (int t = 0; t < 16; ++t) {
    int c = lane + t * 32;
    qb[c] = f2bf(vals[t] * rs * nw[c]);
  }
}

// ---------------------------------------------------------------------------
// Host-side orchestration
// ---------------------------------------------------------------------------
struct MambaP {
  const float *in_proj, *conv_w, *conv_b, *dt_bias, *A_log, *D, *norm_w, *out_proj;
};
struct Bump {
  char* base; size_t off;
  float*  f(long n) { float*  p = (float*)(base + off);  off = (off + n * 4 + 255) & ~255ull; return p; }
  __bf16* h(long n) { __bf16* p = (__bf16*)(base + off); off = (off + n * 2 + 255) & ~255ull; return p; }
};

static void run_mamba(const __bf16* u, float* out, const MambaP& p,
                      char* scratch, hipStream_t s) {
  const int G = GCHUNKS;
  Bump w{scratch, 0};
  __bf16* inpp  = w.h((long)ZLD * DMODEL);      // padded in_proj
  __bf16* outWp = w.h((long)DMODEL * DINNER);   // out_proj bf16
  float*  zx    = w.f((long)LPAD * ZLD);
  float*  xbc   = w.f((long)LPAD * DXBC);
  __bf16* xbcb  = w.h((long)LPAD * DXBC);
  float*  dt    = w.f((long)LPAD * NHEADS);
  float*  acsT  = w.f((long)LPAD * NHEADS);
  float*  dstT  = w.f((long)LPAD * NHEADS);
  __bf16* xdtT  = w.h((long)LPAD * DINNER);
  __bf16* xds   = w.h((long)LPAD * DINNER);
  __bf16* BT    = w.h((long)LPAD * DSTATE);
  float*  CB    = w.f((long)G * 65536);
  float*  st    = w.f((long)G * 65536);
  __bf16* sprev = w.h((long)G * 65536);
  float*  y     = w.f((long)LPAD * DINNER);
  __bf16* ybf   = w.h((long)LPAD * DINNER);

  dim3 blk(128);
  k_padw<<<dim3((ZLD * DMODEL + 255) / 256, 1), 256, 0, s>>>(
      p.in_proj, DMODEL, 1, 0, inpp, ZLD, DMODEL, 0, DINPROJ, DMODEL);
  k_padw<<<dim3((DMODEL * DINNER + 255) / 256, 1), 256, 0, s>>>(
      p.out_proj, DINNER, 1, 0, outWp, DMODEL, DINNER, 0, DMODEL, DINNER);
  // zxbcdt = u @ in_proj^T  (fp32 out; N padded to 1312, zero cols)
  k_gemm<<<dim3(ZLD / 32, LPAD / 128), blk, 0, s>>>(
      u, DMODEL, 0, inpp, DMODEL, 0, nullptr,
      zx, ZLD, 0, LPAD, ZLD, DMODEL, 0, 0);
  k_dt<<<(LPAD * NHEADS + 255) / 256, 256, 0, s>>>(zx, p.dt_bias, dt);
  k_convseq<<<(unsigned)(((long)LPAD * DXBC + 255) / 256), 256, 0, s>>>(
      zx, p.conv_w, p.conv_b, xbc, xbcb);
  k_cumsum<<<G * NHEADS, CHUNK, 0, s>>>(dt, p.A_log, acsT);
  k_dstate<<<(LPAD * NHEADS + 255) / 256, 256, 0, s>>>(acsT, dstT);
  k_xprep<<<(unsigned)(((long)LPAD * DINNER + 255) / 256), 256, 0, s>>>(
      xbc, dt, dstT, xdtT, xds);
  k_BT<<<(unsigned)(((long)LPAD * DSTATE + 255) / 256), 256, 0, s>>>(xbc, BT);
  // CB[g] = C_g @ B_g^T  (batched over chunks; fp32 out)
  k_gemm<<<dim3(CHUNK / 32, 2, G), blk, 0, s>>>(
      xbcb + DINNER + DSTATE, DXBC, (long)CHUNK * DXBC,
      xbcb + DINNER, DXBC, (long)CHUNK * DXBC, nullptr,
      CB, CHUNK, 65536, CHUNK, CHUNK, DSTATE, 0, 0);
  k_ydiag<<<dim3(4, 4, G * NHEADS), blk, 0, s>>>(CB, acsT, xdtT, y);
  k_states<<<dim3(2, 4, G * NHEADS), blk, 0, s>>>(xds, BT, st);
  k_scan<<<256, 256, 0, s>>>(st, acsT, sprev);
  k_yoff<<<dim3(4, 4, G * NHEADS), blk, 0, s>>>(xbcb, xbc, sprev, acsT, p.D, y);
  k_gatenorm<<<LSEQ, 32, 0, s>>>(y, zx, p.norm_w, ybf);
  k_gemm<<<dim3(DMODEL / 32, LPAD / 128), blk, 0, s>>>(
      ybf, DINNER, 0, outWp, DINNER, 0, nullptr,
      out, DMODEL, 0, LPAD, DMODEL, DINNER, 0, 0);
}

extern "C" void kernel_launch(void* const* d_in, const int* in_sizes, int n_in,
                              void* d_out, int out_size, void* d_ws, size_t ws_size,
                              hipStream_t stream) {
  (void)in_sizes; (void)n_in; (void)out_size; (void)ws_size;
  const float* x         = (const float*)d_in[0];
  const float* ln_pre_g  = (const float*)d_in[1];
  const float* ln_pre_b  = (const float*)d_in[2];
  const float* sp_W      = (const float*)d_in[3];
  const float* sp_b      = (const float*)d_in[4];
  MambaP mspec { (const float*)d_in[5],  (const float*)d_in[6],  (const float*)d_in[7],
                 (const float*)d_in[8],  (const float*)d_in[9],  (const float*)d_in[10],
                 (const float*)d_in[11], (const float*)d_in[12] };
  const float* spec_ln_g = (const float*)d_in[13];
  const float* spec_ln_b = (const float*)d_in[14];
  const float* conv_W    = (const float*)d_in[15];
  const float* conv_b    = (const float*)d_in[16];
  MambaP mspat { (const float*)d_in[17], (const float*)d_in[18], (const float*)d_in[19],
                 (const float*)d_in[20], (const float*)d_in[21], (const float*)d_in[22],
                 (const float*)d_in[23], (const float*)d_in[24] };
  const float* spat_ln_g = (const float*)d_in[25];
  const float* spat_ln_b = (const float*)d_in[26];
  const float* fus_W     = (const float*)d_in[27];
  const float* fus_b     = (const float*)d_in[28];
  const float* fus_ln_g  = (const float*)d_in[29];
  const float* fus_ln_b  = (const float*)d_in[30];
  MambaP mfus  { (const float*)d_in[31], (const float*)d_in[32], (const float*)d_in[33],
                 (const float*)d_in[34], (const float*)d_in[35], (const float*)d_in[36],
                 (const float*)d_in[37], (const float*)d_in[38] };
  const float* fusn_ln_g = (const float*)d_in[39];
  const float* fusn_ln_b = (const float*)d_in[40];
  const float* c1_W      = (const float*)d_in[41];
  const float* c1_b      = (const float*)d_in[42];
  const float* c2_W      = (const float*)d_in[43];
  const float* c2_b      = (const float*)d_in[44];
  const float* c3_W      = (const float*)d_in[45];
  const float* c3_b      = (const float*)d_in[46];

  Bump w{(char*)d_ws, 0};
  __bf16* xn      = w.h((long)LPAD * KB_PAD);     // bf16 activations
  __bf16* xp      = w.h((long)LPAD * DMODEL);
  __bf16* comb    = w.h((long)LPAD * 2 * DMODEL);
  __bf16* spat_in = w.h((long)LPAD * DMODEL);
  __bf16* fus     = w.h((long)LPAD * DMODEL);
  __bf16* fusn    = w.h((long)LPAD * DMODEL);
  __bf16* h1      = w.h((long)LPAD * KB_PAD);
  __bf16* h2      = w.h((long)LPAD * DSTATE);
  float*  mo      = w.f((long)LPAD * DMODEL);     // fp32 pre-norm temp
  __bf16* spWp    = w.h(DMODEL * KB_PAD);
  __bf16* cWT     = w.h(9 * DMODEL * KB_PAD);
  __bf16* c1Wp    = w.h(KB_PAD * DMODEL);
  __bf16* c2Wp    = w.h(DSTATE * KB_PAD);
  __bf16* c3Wp    = w.h(32 * DSTATE);
  __bf16* fusWp   = w.h(DMODEL * 2 * DMODEL);
  float*  c1bp    = w.f(KB_PAD);
  float*  c3bp    = w.f(32);
  char*   mscr    = (char*)d_ws + w.off;          // shared mamba scratch

  dim3 blk(128);
  // --- weight pad/transpose copies to bf16 (tiny) ---
  k_padw<<<dim3((DMODEL * KB_PAD + 255) / 256, 1), 256, 0, stream>>>(
      sp_W, BANDS, 1, 0, spWp, DMODEL, KB_PAD, 0, DMODEL, BANDS);
  k_padw<<<dim3((DMODEL * KB_PAD + 255) / 256, 9), 256, 0, stream>>>(
      conv_W, 1, DMODEL, (long)BANDS * DMODEL,
      cWT, DMODEL, KB_PAD, (long)DMODEL * KB_PAD, DMODEL, BANDS);
  k_padw<<<dim3((KB_PAD * DMODEL + 255) / 256, 1), 256, 0, stream>>>(
      c1_W, DMODEL, 1, 0, c1Wp, KB_PAD, DMODEL, 0, 200, DMODEL);
  k_padb<<<1, KB_PAD, 0, stream>>>(c1_b, c1bp, 200, KB_PAD);
  k_padw<<<dim3((DSTATE * KB_PAD + 255) / 256, 1), 256, 0, stream>>>(
      c2_W, BANDS, 1, 0, c2Wp, DSTATE, KB_PAD, 0, DSTATE, BANDS);
  k_padw<<<dim3((32 * DSTATE + 255) / 256, 1), 256, 0, stream>>>(
      c3_W, DSTATE, 1, 0, c3Wp, 32, DSTATE, 0, NCLS, DSTATE);
  k_padb<<<1, 32, 0, stream>>>(c3_b, c3bp, NCLS, 32);
  k_padw<<<dim3((DMODEL * 2 * DMODEL + 255) / 256, 1), 256, 0, stream>>>(
      fus_W, 2 * DMODEL, 1, 0, fusWp, DMODEL, 2 * DMODEL, 0, DMODEL, 2 * DMODEL);

  // 1) band LayerNorm -> bf16 xn
  k_ln<<<LSEQ, 32, 0, stream>>>(x, BANDS, ln_pre_g, ln_pre_b,
                                xn, KB_PAD, 0, LSEQ, BANDS, 0);
  // 2) spectral projection + GELU -> bf16 xp
  k_gemm<<<dim3(DMODEL / 32, LPAD / 128), blk, 0, stream>>>(
      xn, KB_PAD, 0, spWp, KB_PAD, 0, sp_b,
      xp, DMODEL, 0, LPAD, DMODEL, KB_PAD, 1, 1);
  // 3) spectral Mamba2 -> gelu(ln) into comb[:, 0:256]
  run_mamba(xp, mo, mspec, mscr, stream);
  k_ln<<<LSEQ, 32, 0, stream>>>(mo, DMODEL, spec_ln_g, spec_ln_b,
                                comb, 2 * DMODEL, 0, LSEQ, DMODEL, 1);
  // 4) 3x3 conv (9 taps fused) + bias + GELU -> bf16 spat_in
  k_conv<<<dim3(DMODEL / 32, LPAD / 128), blk, 0, stream>>>(xn, cWT, conv_b, spat_in);
  // 5) spatial Mamba2 -> gelu(ln) into comb[:, 256:512]
  run_mamba(spat_in, mo, mspat, mscr, stream);
  k_ln<<<LSEQ, 32, 0, stream>>>(mo, DMODEL, spat_ln_g, spat_ln_b,
                                comb, 2 * DMODEL, DMODEL, LSEQ, DMODEL, 1);
  // 6) fusion GEMM (fp32 out) then gelu(ln) -> bf16 fus
  k_gemm<<<dim3(DMODEL / 32, LPAD / 128), blk, 0, stream>>>(
      comb, 2 * DMODEL, 0, fusWp, 2 * DMODEL, 0, fus_b,
      mo, DMODEL, 0, LPAD, DMODEL, 2 * DMODEL, 0, 0);
  k_ln<<<LSEQ, 32, 0, stream>>>(mo, DMODEL, fus_ln_g, fus_ln_b,
                                fus, DMODEL, 0, LSEQ, DMODEL, 1);
  // 7) fused Mamba2 -> gelu(ln) -> bf16 fusn
  run_mamba(fus, mo, mfus, mscr, stream);
  k_ln<<<LSEQ, 32, 0, stream>>>(mo, DMODEL, fusn_ln_g, fusn_ln_b,
                                fusn, DMODEL, 0, LSEQ, DMODEL, 1);
  // 8) classifier head (h1, h2 bf16; final fp32 to d_out)
  k_gemm<<<dim3(KB_PAD / 32, LPAD / 128), blk, 0, stream>>>(
      fusn, DMODEL, 0, c1Wp, DMODEL, 0, c1bp,
      h1, KB_PAD, 0, LPAD, KB_PAD, DMODEL, 1, 1);
  k_gemm<<<dim3(DSTATE / 32, LPAD / 128), blk, 0, stream>>>(
      h1, KB_PAD, 0, c2Wp, KB_PAD, 0, c2_b,
      h2, DSTATE, 0, LPAD, DSTATE, KB_PAD, 1, 1);
  k_gemm<<<dim3(1, LPAD / 128), blk, 0, stream>>>(
      h2, DSTATE, 0, c3Wp, DSTATE, 0, c3bp,
      d_out, NCLS, 0, LSEQ, NCLS, DSTATE, 0, 0);
}
